// S4Layer_24850680774869
// MI455X (gfx1250) — matrix operand order repaired and checked by reading, not compiled, not run on hardware
//
#include <hip/hip_runtime.h>

#define NSTATE 64
#define LSEQ   2048
#define ROWS   8192          // BATCH*H = 16*512
#define STEPF  0.001f
#define BM 128
#define BN 128
#define BK 32
#define LDSS 48              // padded LDS row stride (bf16 elems), 96B keeps 16B alignment

typedef __attribute__((ext_vector_type(16))) __bf16 v16bf;
typedef __attribute__((ext_vector_type(8)))  float  v8f;

#if __has_builtin(__builtin_amdgcn_global_load_async_to_lds_b128)
#define HAVE_ASYNC_LDS 1
#define AS1 __attribute__((address_space(1)))
#define AS3 __attribute__((address_space(3)))
typedef int vs4i __attribute__((vector_size(4 * sizeof(int))));  // matches builtin's param pointee
#endif

__device__ __forceinline__ void wait_async_zero() {
#if defined(HAVE_ASYNC_LDS)
#if __has_builtin(__builtin_amdgcn_s_wait_asynccnt)
  __builtin_amdgcn_s_wait_asynccnt(0);
#else
  asm volatile("s_wait_asynccnt 0" ::: "memory");
#endif
#endif
}

// ---------------- complex helpers (f32) ----------------
struct c32 { float x, y; };
__device__ __forceinline__ c32 cmul(c32 a, c32 b){ return {a.x*b.x - a.y*b.y, a.x*b.y + a.y*b.x}; }
__device__ __forceinline__ c32 cadd(c32 a, c32 b){ return {a.x+b.x, a.y+b.y}; }
__device__ __forceinline__ c32 csub(c32 a, c32 b){ return {a.x-b.x, a.y-b.y}; }
__device__ __forceinline__ c32 cinv(c32 a){ float d = 1.0f/(a.x*a.x + a.y*a.y); return {a.x*d, -a.y*d}; }

// ---------------- bf16 split helpers ----------------
__device__ __forceinline__ unsigned short f2bf_rne(float f) {
  unsigned u = __float_as_uint(f);
  return (unsigned short)((u + 0x7FFFu + ((u >> 16) & 1u)) >> 16);
}
__device__ __forceinline__ float bf2f(unsigned short h) {
  return __uint_as_float(((unsigned)h) << 16);
}

// ---------------- stage 1: generating-function kernel K[j] ----------------
__global__ void s4_cauchy_kernel(const float* __restrict__ Lre, const float* __restrict__ Lim,
                                 const float* __restrict__ Pre, const float* __restrict__ Pim,
                                 const float* __restrict__ Bre, const float* __restrict__ Bim,
                                 const float* __restrict__ Cre, const float* __restrict__ Cim,
                                 float* __restrict__ Kre, float* __restrict__ Kim) {
  int j = blockIdx.x * blockDim.x + threadIdx.x;
  if (j >= LSEQ) return;
  float th = (float)j * (6.28318530717958647692f / (float)LSEQ);
  float sn, cs; __sincosf(th, &sn, &cs);
  c32 om   = {cs, -sn};                      // exp(-2*pi*i*j/L)
  c32 den1 = {1.0f + om.x, om.y};            // 1 + omega
  c32 id1  = cinv(den1);
  c32 g    = cmul({2.0f / STEPF, 0.0f}, cmul({1.0f - om.x, -om.y}, id1));
  c32 cf   = cmul({2.0f, 0.0f}, id1);
  c32 k00 = {0,0}, k01 = {0,0}, k10 = {0,0}, k11 = {0,0};
  for (int n = 0; n < NSTATE; ++n) {
    c32 r  = cinv({g.x - Lre[n], g.y - Lim[n]});   // Cauchy term 1/(g - lambda)
    c32 a0 = {Cre[n], -Cim[n]};                    // conj(C)
    c32 a1 = {Pre[n], -Pim[n]};                    // conj(P)
    c32 b0 = {Bre[n],  Bim[n]};
    c32 b1 = {Pre[n],  Pim[n]};
    k00 = cadd(k00, cmul(r, cmul(a0, b0)));
    k01 = cadd(k01, cmul(r, cmul(a0, b1)));
    k10 = cadd(k10, cmul(r, cmul(a1, b0)));
    k11 = cadd(k11, cmul(r, cmul(a1, b1)));
  }
  c32 t  = cmul(cmul(k01, k10), cinv({1.0f + k11.x, k11.y}));
  c32 kv = cmul(cf, csub(k00, t));
  Kre[j] = kv.x; Kim[j] = kv.y;
}

// ---------------- stage 2: k_re = real(ifft(K)), direct DFT (tiny) ----------------
__global__ void s4_idft_kernel(const float* __restrict__ Kre, const float* __restrict__ Kim,
                               float* __restrict__ kre) {
  int t = blockIdx.x * blockDim.x + threadIdx.x;
  if (t >= LSEQ) return;
  const float c0 = 6.28318530717958647692f / (float)LSEQ;
  float acc = 0.0f;
  for (int j = 0; j < LSEQ; ++j) {
    int m = (j * t) & (LSEQ - 1);              // exact angle reduction mod L
    float ang = (float)m * c0;
    float s, c; __sincosf(ang, &s, &c);
    acc += Kre[j] * c - Kim[j] * s;            // real part of K[j]*exp(+2*pi*i*j*t/L)
  }
  kre[t] = acc * (1.0f / (float)LSEQ);
}

// ---- stage 3: transposed circulant, pre-split to bf16 hi/lo ----
// CThi/CTlo[n][k] = split(k_re[(n-k) mod L]) — final WMMA bit-layout, so the GEMM
// can DMA it straight into LDS with async loads (no per-iteration conversion).
__global__ void s4_fillct_kernel(const float* __restrict__ kre,
                                 unsigned short* __restrict__ CThi,
                                 unsigned short* __restrict__ CTlo) {
  int idx = blockIdx.x * blockDim.x + threadIdx.x;
  if (idx >= LSEQ * LSEQ) return;
  int n = idx >> 11;
  int k = idx & (LSEQ - 1);
  float v = kre[(n - k) & (LSEQ - 1)];
  unsigned short h = f2bf_rne(v);
  CThi[idx] = h;
  CTlo[idx] = f2bf_rne(v - bf2f(h));
}

// load 16 f32 from global, split each into bf16 hi + bf16 lo (Markidis split)
__device__ __forceinline__ void load_split16(const float* __restrict__ gp,
                                             uint4* hi, uint4* lo) {
  float s[16];
  *(float4*)(s + 0)  = ((const float4*)gp)[0];
  *(float4*)(s + 4)  = ((const float4*)gp)[1];
  *(float4*)(s + 8)  = ((const float4*)gp)[2];
  *(float4*)(s + 12) = ((const float4*)gp)[3];
  unsigned h[8], l[8];
#pragma unroll
  for (int i = 0; i < 8; ++i) {
    unsigned short ha = f2bf_rne(s[2*i]);
    unsigned short hb = f2bf_rne(s[2*i+1]);
    unsigned short la = f2bf_rne(s[2*i]   - bf2f(ha));
    unsigned short lb = f2bf_rne(s[2*i+1] - bf2f(hb));
    h[i] = (unsigned)ha | ((unsigned)hb << 16);
    l[i] = (unsigned)la | ((unsigned)lb << 16);
  }
  hi[0].x = h[0]; hi[0].y = h[1]; hi[0].z = h[2]; hi[0].w = h[3];
  hi[1].x = h[4]; hi[1].y = h[5]; hi[1].z = h[6]; hi[1].w = h[7];
  lo[0].x = l[0]; lo[0].y = l[1]; lo[0].z = l[2]; lo[0].w = l[3];
  lo[1].x = l[4]; lo[1].y = l[5]; lo[1].z = l[6]; lo[1].w = l[7];
}

__device__ __forceinline__ v16bf mkfrag(uint4 a, uint4 b) {
  union { v16bf v; uint4 q[2]; } u;
  u.q[0] = a; u.q[1] = b;
  return u.v;
}

// A fragment, ISA 16-bit A 16x32 layout:
//   lanes 0-15 (M=lane):     elems 0..7 -> K 0..7,   elems 8..15 -> K 16..23
//   lanes 16-31 (M=lane-16): elems 0..7 -> K 8..15,  elems 8..15 -> K 24..31
__device__ __forceinline__ v16bf ld_fragA(const unsigned short* tile, int rowBase, int lane) {
  int r = lane & 15, h = lane >> 4;
  const unsigned short* p = tile + (rowBase + r) * LDSS + h * 8;
  return mkfrag(*(const uint4*)p, *(const uint4*)(p + 16));
}
// B fragment (B rows striped across lanes): lane n (0-15) = column n, K 0..7 and 16..23;
// lane n+16 = column n, K 8..15 and 24..31  -> same K order as the A fragment.
__device__ __forceinline__ v16bf ld_fragB(const unsigned short* tile, int rowBase, int lane) {
  int r = lane & 15, h = lane >> 4;
  const unsigned short* p = tile + (rowBase + r) * LDSS + h * 8;
  return mkfrag(*(const uint4*)p, *(const uint4*)(p + 16));
}

// ---------------- stage 4: Y[8192,2048] = X @ circulant, split-bf16 WMMA ----------------
__global__ void __launch_bounds__(256)
s4_gemm_kernel(const float* __restrict__ x,
               const unsigned short* __restrict__ CThi,
               const unsigned short* __restrict__ CTlo,
               float* __restrict__ y) {
  __shared__ unsigned short Ahi[BM * LDSS];
  __shared__ unsigned short Alo[BM * LDSS];
  __shared__ unsigned short Bhi[BN * LDSS];
  __shared__ unsigned short Blo[BN * LDSS];

  const int t    = threadIdx.x;
  const int lane = t & 31;
  const int wave = t >> 5;        // 0..7
  const int wm   = wave >> 1;     // 0..3 -> 32-row slab
  const int wn   = wave & 1;      // 0..1 -> 64-col slab
  const int rowM0 = blockIdx.y * BM;
  const int col0  = blockIdx.x * BN;
  const int srow = t >> 1;        // staging row 0..127
  const int sseg = (t & 1) << 4;  // staging K segment 0 or 16

  v8f acc[2][4] = {};

  for (int k0 = 0; k0 < LSEQ; k0 += BK) {
    __syncthreads();              // previous iteration's fragment reads done
    {
      // ---- A tile: f32 -> split bf16 (must pass through VALU) ----
      uint4 h[2], l[2];
      load_split16(x + (size_t)(rowM0 + srow) * LSEQ + (k0 + sseg), h, l);
      *(uint4*)(Ahi + srow * LDSS + sseg)     = h[0];
      *(uint4*)(Ahi + srow * LDSS + sseg + 8) = h[1];
      *(uint4*)(Alo + srow * LDSS + sseg)     = l[0];
      *(uint4*)(Alo + srow * LDSS + sseg + 8) = l[1];

      // ---- B tile: already bf16 in memory -> async DMA global->LDS ----
      const size_t goff = (size_t)(col0 + srow) * LSEQ + (k0 + sseg); // bf16 elems
      const int    loff = srow * LDSS + sseg;                         // bf16 elems
#if defined(HAVE_ASYNC_LDS)
      // INST_OFFSET is added to both the global and LDS address (ISA 08 §4.4),
      // so one address pair covers each 32B chunk with imm offsets 0 and 16.
      __builtin_amdgcn_global_load_async_to_lds_b128(
          (AS1 vs4i*)(CThi + goff), (AS3 vs4i*)(Bhi + loff), 0, 0);
      __builtin_amdgcn_global_load_async_to_lds_b128(
          (AS1 vs4i*)(CThi + goff), (AS3 vs4i*)(Bhi + loff), 16, 0);
      __builtin_amdgcn_global_load_async_to_lds_b128(
          (AS1 vs4i*)(CTlo + goff), (AS3 vs4i*)(Blo + loff), 0, 0);
      __builtin_amdgcn_global_load_async_to_lds_b128(
          (AS1 vs4i*)(CTlo + goff), (AS3 vs4i*)(Blo + loff), 16, 0);
#else
      {
        uint4 b0 = *(const uint4*)(CThi + goff);
        uint4 b1 = *(const uint4*)(CThi + goff + 8);
        uint4 c0 = *(const uint4*)(CTlo + goff);
        uint4 c1 = *(const uint4*)(CTlo + goff + 8);
        *(uint4*)(Bhi + loff)     = b0;
        *(uint4*)(Bhi + loff + 8) = b1;
        *(uint4*)(Blo + loff)     = c0;
        *(uint4*)(Blo + loff + 8) = c1;
      }
#endif
      if (k0 + BK < LSEQ)
        __builtin_prefetch(x + (size_t)(rowM0 + srow) * LSEQ + (k0 + BK + sseg), 0, 0);
    }
    wait_async_zero();            // own wave's async B-tile DMA landed in LDS
    __syncthreads();              // everyone's staging visible

    v16bf ah[2], al[2], bh[4], bl[4];
#pragma unroll
    for (int mt = 0; mt < 2; ++mt) {
      int rb = wm * 32 + mt * 16;
      ah[mt] = ld_fragA(Ahi, rb, lane);
      al[mt] = ld_fragA(Alo, rb, lane);
    }
#pragma unroll
    for (int nt = 0; nt < 4; ++nt) {
      int rb = wn * 64 + nt * 16;
      bh[nt] = ld_fragB(Bhi, rb, lane);
      bl[nt] = ld_fragB(Blo, rb, lane);
    }
#pragma unroll
    for (int mt = 0; mt < 2; ++mt) {
#pragma unroll
      for (int nt = 0; nt < 4; ++nt) {
        v8f c = acc[mt][nt];
        c = __builtin_amdgcn_wmma_f32_16x16x32_bf16(false, ah[mt], false, bh[nt], (short)0, c, false, false);
        c = __builtin_amdgcn_wmma_f32_16x16x32_bf16(false, ah[mt], false, bl[nt], (short)0, c, false, false);
        c = __builtin_amdgcn_wmma_f32_16x16x32_bf16(false, al[mt], false, bh[nt], (short)0, c, false, false);
        acc[mt][nt] = c;
      }
    }
  }

  // Epilogue: C/D layout VGPR i -> M = i (lanes 0-15), M = 8+i (lanes 16-31); N = lane&15.
#pragma unroll
  for (int mt = 0; mt < 2; ++mt) {
#pragma unroll
    for (int nt = 0; nt < 4; ++nt) {
      int row0 = rowM0 + wm * 32 + mt * 16 + (lane >> 4) * 8;
      int col  = col0 + wn * 64 + nt * 16 + (lane & 15);
#pragma unroll
      for (int i = 0; i < 8; ++i)
        y[(size_t)(row0 + i) * LSEQ + col] = acc[mt][nt][i];
    }
  }
}

extern "C" void kernel_launch(void* const* d_in, const int* in_sizes, int n_in,
                              void* d_out, int out_size, void* d_ws, size_t ws_size,
                              hipStream_t stream) {
  const float* x   = (const float*)d_in[0];
  const float* Lre = (const float*)d_in[1];
  const float* Lim = (const float*)d_in[2];
  const float* Pre = (const float*)d_in[3];
  const float* Pim = (const float*)d_in[4];
  const float* Bre = (const float*)d_in[5];
  const float* Bim = (const float*)d_in[6];
  const float* Cre = (const float*)d_in[7];
  const float* Cim = (const float*)d_in[8];
  float* y = (float*)d_out;

  // Workspace: [Kre 8KB][Kim 8KB][k_re 8KB][pad to 64KB][CThi 8MB][CTlo 8MB]
  float* Kre = (float*)d_ws;
  float* Kim = Kre + LSEQ;
  float* kre = Kim + LSEQ;
  unsigned short* CThi = (unsigned short*)((char*)d_ws + (64u << 10));
  unsigned short* CTlo = CThi + (size_t)LSEQ * LSEQ;

  s4_cauchy_kernel<<<LSEQ / 256, 256, 0, stream>>>(Lre, Lim, Pre, Pim, Bre, Bim, Cre, Cim, Kre, Kim);
  s4_idft_kernel<<<LSEQ / 256, 256, 0, stream>>>(Kre, Kim, kre);
  s4_fillct_kernel<<<(LSEQ * LSEQ) / 256, 256, 0, stream>>>(kre, CThi, CTlo);
  dim3 grid(LSEQ / BN, ROWS / BM);
  s4_gemm_kernel<<<grid, 256, 0, stream>>>(x, CThi, CTlo, y);
}
